// SpatialReasoningModule_70377334112318
// MI455X (gfx1250) — matrix operand order repaired and checked by reading, not compiled, not run on hardware
//
#include <hip/hip_runtime.h>
#include <hip/hip_bf16.h>
#include <math.h>

typedef __attribute__((ext_vector_type(16))) _Float16 v16h;
typedef __attribute__((ext_vector_type(8)))  float    v8f;

#define Bn 1
#define Nn 1024
#define Dn 128
#define NH 8
#define HD 16

__device__ __forceinline__ float gelu_exact(float v) {
    return 0.5f * v * (1.0f + erff(v * 0.70710678118654752440f));
}

// ---------------------------------------------------------------------------
// conv1 (1->32 ch, 3x3, pad 1) + exact GELU.  grid=32 (out channel), block=1024 (pixel)
// ---------------------------------------------------------------------------
__global__ void conv1_gelu_kernel(const float* __restrict__ dm,
                                  const float* __restrict__ w,
                                  const float* __restrict__ b,
                                  float* __restrict__ t1) {
    const int c = blockIdx.x;
    const int p = threadIdx.x;
    const int y = p >> 5, x = p & 31;
    float s = b[c];
#pragma unroll
    for (int dy = -1; dy <= 1; ++dy)
#pragma unroll
        for (int dx = -1; dx <= 1; ++dx) {
            int yy = y + dy, xx = x + dx;
            if (yy >= 0 && yy < 32 && xx >= 0 && xx < 32)
                s += dm[yy * 32 + xx] * w[c * 9 + (dy + 1) * 3 + (dx + 1)];
        }
    t1[c * 1024 + p] = gelu_exact(s);
}

// ---------------------------------------------------------------------------
// conv2 (32->128 ch, 3x3, pad 1) fused with x = features + 0.1*dep_flat
// grid=128 (out channel), block=1024 (pixel)
// ---------------------------------------------------------------------------
__global__ void conv2_x_kernel(const float* __restrict__ t1,
                               const float* __restrict__ w,
                               const float* __restrict__ b,
                               const float* __restrict__ feat,
                               float* __restrict__ x) {
    const int c = blockIdx.x;
    const int p = threadIdx.x;
    const int y = p >> 5, xx = p & 31;
    float s = b[c];
    for (int ic = 0; ic < 32; ++ic) {
        const float* wp = w + (c * 32 + ic) * 9;
        const float* tp = t1 + ic * 1024;
#pragma unroll
        for (int dy = -1; dy <= 1; ++dy)
#pragma unroll
            for (int dx = -1; dx <= 1; ++dx) {
                int yy = y + dy, xc = xx + dx;
                if (yy >= 0 && yy < 32 && xc >= 0 && xc < 32)
                    s += tp[yy * 32 + xc] * wp[(dy + 1) * 3 + (dx + 1)];
            }
    }
    x[p * Dn + c] = feat[p * Dn + c] + 0.1f * s;
}

// ---------------------------------------------------------------------------
// WMMA GEMM, K=128: out[m,n] = sum_k A[m,k] * W[n, koff+k] + bias[n]
// One wave computes a 16x64 tile: A-fragment reused across 4 B-fragments,
// 4 independent accumulator chains, K-loop fully unrolled -> 16 v_wmma
// per wave with back-to-back XDL issue.
// grid = (Nout/64, M/16), block = 32.
// ---------------------------------------------------------------------------
__global__ void __launch_bounds__(32)
wmma_gemm_k128(const float* __restrict__ A, const float* __restrict__ W,
               int ldw, int koff, const float* __restrict__ bias,
               float* __restrict__ out, int ldo) {
    const int lane = threadIdx.x & 31;
    const int lo = lane & 15;
    const int hi = lane >> 4;
    const int nBase = blockIdx.x * 64;
    const int mBase = blockIdx.y * 16;

    const float* arow = A + (mBase + lo) * 128; // A-frag: row M = lo

    v8f acc[4] = {{}, {}, {}, {}};
#pragma unroll
    for (int kc = 0; kc < 128; kc += 32) {
        // A fragment (layout: K = h + 8*(h>=8) + 8*(lane>=16)), loaded once
        v16h a;
#pragma unroll
        for (int h = 0; h < 16; ++h) {
            int ka = kc + h + ((h & 8) ? 8 : 0) + (hi ? 8 : 0);
            a[h] = (_Float16)arow[ka];
        }
#pragma unroll
        for (int nt = 0; nt < 4; ++nt) {
            const float* wrow = W + (nBase + nt * 16 + lo) * ldw + koff;
            v16h bf;
#pragma unroll
            for (int h = 0; h < 16; ++h) {
                int kb = kc + h + ((h & 8) ? 8 : 0) + (hi ? 8 : 0); // B layout: K = h + 8*(h>=8) + 8*(lane>=16)
                bf[h] = (_Float16)wrow[kb];
            }
            acc[nt] = __builtin_amdgcn_wmma_f32_16x16x32_f16(
                false, a, false, bf, (short)0, acc[nt], false, false);
        }
    }
#pragma unroll
    for (int nt = 0; nt < 4; ++nt) {
        const int n = nBase + nt * 16 + lo;
        const float bv = bias ? bias[n] : 0.0f;
#pragma unroll
        for (int r = 0; r < 8; ++r)
            out[(mBase + r + 8 * hi) * ldo + n] = acc[nt][r] + bv;
    }
}

// ---------------------------------------------------------------------------
// Flash attention, one wave per (q-tile 16 rows, head). HD=16, K padded to 32.
// grid = (64 q-tiles, 8 heads), block = 32.
// ---------------------------------------------------------------------------
__global__ void __launch_bounds__(32)
flash_attn_kernel(const float* __restrict__ qkv, float* __restrict__ o) {
    __shared__ float pS[256];
    const int lane = threadIdx.x & 31;
    const int lo = lane & 15;
    const int hi = lane >> 4;
    const int qBase = blockIdx.x * 16;
    const int head  = blockIdx.y;
    const int qoff = head * HD;
    const int koff = Dn + head * HD;
    const int voff = 2 * Dn + head * HD;

    // Q fragment (A layout), zero-padded K 16..31
    v16h qa;
#pragma unroll
    for (int h = 0; h < 16; ++h) {
        int k = h + ((h & 8) ? 8 : 0) + (hi ? 8 : 0);
        qa[h] = (k < HD) ? (_Float16)qkv[(qBase + lo) * 384 + qoff + k]
                         : (_Float16)0.0f;
    }

    float rowmax[8], rowsum[8];
    v8f acc = {};
#pragma unroll
    for (int r = 0; r < 8; ++r) { rowmax[r] = -3.0e38f; rowsum[r] = 0.0f; }

    for (int jt = 0; jt < 64; ++jt) {
        const int jBase = jt * 16;
        // K fragment (B layout): element (k=d, n=j); pad d>=16 with 0
        v16h kb;
#pragma unroll
        for (int h = 0; h < 16; ++h) {
            int k = h + ((h & 8) ? 8 : 0) + (hi ? 8 : 0);
            kb[h] = (k < HD) ? (_Float16)qkv[(jBase + lo) * 384 + koff + k]
                             : (_Float16)0.0f;
        }
        v8f s = {};
        s = __builtin_amdgcn_wmma_f32_16x16x32_f16(false, qa, false, kb,
                                                   (short)0, s, false, false);
        // online softmax; rows live within a 16-lane half
#pragma unroll
        for (int r = 0; r < 8; ++r) {
            float sv = s[r] * 0.25f; // 1/sqrt(16)
            float mx = sv;
#pragma unroll
            for (int off = 8; off > 0; off >>= 1)
                mx = fmaxf(mx, __shfl_xor(mx, off, 16));
            float mnew = fmaxf(rowmax[r], mx);
            float alpha = __expf(rowmax[r] - mnew);
            float p = __expf(sv - mnew);
            float ps = p;
#pragma unroll
            for (int off = 8; off > 0; off >>= 1)
                ps += __shfl_xor(ps, off, 16);
            rowsum[r] = rowsum[r] * alpha + ps;
            rowmax[r] = mnew;
            acc[r] *= alpha;
            pS[(r + 8 * hi) * 16 + lo] = p; // P tile row-major in LDS
        }
        __syncthreads();
        // P as A fragment (from LDS), V as B fragment; K padded 16->32
        v16h pa, vb;
#pragma unroll
        for (int h = 0; h < 16; ++h) {
            int ka = h + ((h & 8) ? 8 : 0) + (hi ? 8 : 0);
            pa[h] = (ka < 16) ? (_Float16)pS[lo * 16 + ka] : (_Float16)0.0f;
            int kv = h + ((h & 8) ? 8 : 0) + (hi ? 8 : 0);
            vb[h] = (kv < 16) ? (_Float16)qkv[(jBase + kv) * 384 + voff + lo]
                              : (_Float16)0.0f;
        }
        acc = __builtin_amdgcn_wmma_f32_16x16x32_f16(false, pa, false, vb,
                                                     (short)0, acc, false, false);
        __syncthreads();
    }
#pragma unroll
    for (int r = 0; r < 8; ++r)
        o[(qBase + r + 8 * hi) * Dn + head * HD + lo] = acc[r] / rowsum[r];
}

// ---------------------------------------------------------------------------
// hbar[i,d] = (1/1024) * sum_j gelu(row[i,d] + col[j,d] + gc1_b[d])
// grid = 1024 (i), block = 128 (d).  Dominant VALU/erf kernel.
// ---------------------------------------------------------------------------
__global__ void hbar_kernel(const float* __restrict__ rowb,
                            const float* __restrict__ colb,
                            const float* __restrict__ gb,
                            float* __restrict__ hb) {
    const int i = blockIdx.x, d = threadIdx.x;
    const float r = rowb[i * Dn + d] + gb[d];
    float s = 0.0f;
    for (int j = 0; j < Nn; ++j) {
        if (j + 16 < Nn) __builtin_prefetch(colb + (j + 16) * Dn + d, 0, 0);
        s += gelu_exact(r + colb[j * Dn + d]);
    }
    hb[i * Dn + d] = s * (1.0f / (float)Nn);
}

// ---------------------------------------------------------------------------
// y = attended + 0.5*graph_agg, then LayerNorm over D. grid=1024, block=128.
// ---------------------------------------------------------------------------
__global__ void ln_kernel(const float* __restrict__ att,
                          const float* __restrict__ gagg,
                          const float* __restrict__ lw,
                          const float* __restrict__ lb,
                          float* __restrict__ out) {
    __shared__ float red[Dn];
    const int n = blockIdx.x, d = threadIdx.x;
    const float y = att[n * Dn + d] + 0.5f * gagg[n * Dn + d];
    red[d] = y;
    __syncthreads();
    for (int off = 64; off > 0; off >>= 1) {
        if (d < off) red[d] += red[d + off];
        __syncthreads();
    }
    const float mu = red[0] * (1.0f / (float)Dn);
    __syncthreads();
    const float dv = y - mu;
    red[d] = dv * dv;
    __syncthreads();
    for (int off = 64; off > 0; off >>= 1) {
        if (d < off) red[d] += red[d + off];
        __syncthreads();
    }
    const float var = red[0] * (1.0f / (float)Dn);
    out[n * Dn + d] = dv * rsqrtf(var + 1e-5f) * lw[d] + lb[d];
}

// ---------------------------------------------------------------------------
extern "C" void kernel_launch(void* const* d_in, const int* in_sizes, int n_in,
                              void* d_out, int out_size, void* d_ws, size_t ws_size,
                              hipStream_t stream) {
    (void)in_sizes; (void)n_in; (void)out_size; (void)ws_size;
    const float* features   = (const float*)d_in[0];
    const float* depth_map  = (const float*)d_in[1];
    const float* conv1_w    = (const float*)d_in[2];
    const float* conv1_b    = (const float*)d_in[3];
    const float* conv2_w    = (const float*)d_in[4];
    const float* conv2_b    = (const float*)d_in[5];
    const float* in_proj_w  = (const float*)d_in[6];
    const float* in_proj_b  = (const float*)d_in[7];
    const float* out_proj_w = (const float*)d_in[8];
    const float* out_proj_b = (const float*)d_in[9];
    const float* gc1_w      = (const float*)d_in[10];
    const float* gc1_b      = (const float*)d_in[11];
    const float* gc2_w      = (const float*)d_in[12];
    const float* gc2_b      = (const float*)d_in[13];
    const float* ln_w       = (const float*)d_in[14];
    const float* ln_b       = (const float*)d_in[15];
    float* out = (float*)d_out;

    float* ws   = (float*)d_ws;
    float* t1   = ws;                 // 32*1024           = 32768
    float* x    = t1   + 32768;       // 1024*128          = 131072
    float* qkv  = x    + 131072;      // 1024*384          = 393216
    float* ob   = qkv  + 393216;      // 1024*128
    float* att  = ob   + 131072;      // 1024*128
    float* rowb = att  + 131072;      // 1024*128
    float* colb = rowb + 131072;      // 1024*128
    float* hb   = colb + 131072;      // 1024*128
    float* gagg = hb   + 131072;      // 1024*128          (total ~5.4 MB)

    // depth-conv path -> x
    conv1_gelu_kernel<<<32, 1024, 0, stream>>>(depth_map, conv1_w, conv1_b, t1);
    conv2_x_kernel<<<128, 1024, 0, stream>>>(t1, conv2_w, conv2_b, features, x);

    // qkv projection: (1024x128) @ (384x128)^T   (16x64 tiles per wave)
    wmma_gemm_k128<<<dim3(384 / 64, Nn / 16), 32, 0, stream>>>(
        x, in_proj_w, 128, 0, in_proj_b, qkv, 384);

    // multi-head flash attention
    flash_attn_kernel<<<dim3(Nn / 16, NH), 32, 0, stream>>>(qkv, ob);

    // output projection
    wmma_gemm_k128<<<dim3(Dn / 64, Nn / 16), 32, 0, stream>>>(
        ob, out_proj_w, 128, 0, out_proj_b, att, Dn);

    // graph branch: row = F @ wa^T, col = F @ wb^T (wa|wb = gc1_w split)
    wmma_gemm_k128<<<dim3(Dn / 64, Nn / 16), 32, 0, stream>>>(
        features, gc1_w, 256, 0, nullptr, rowb, Dn);
    wmma_gemm_k128<<<dim3(Dn / 64, Nn / 16), 32, 0, stream>>>(
        features, gc1_w, 256, 128, nullptr, colb, Dn);

    // hbar = mean_j gelu(row_i + col_j + b)   (mean hoisted past gc2 matmul)
    hbar_kernel<<<Nn, Dn, 0, stream>>>(rowb, colb, gc1_b, hb);

    // graph_agg = hbar @ gc2_w^T + gc2_b
    wmma_gemm_k128<<<dim3(Dn / 64, Nn / 16), 32, 0, stream>>>(
        hb, gc2_w, 128, 0, gc2_b, gagg, Dn);

    // residual combine + layernorm
    ln_kernel<<<Nn, Dn, 0, stream>>>(att, gagg, ln_w, ln_b, out);
}
